// LayerWiseASP_13700945674785
// MI455X (gfx1250) — matrix operand order repaired and checked by reading, not compiled, not run on hardware
//
#include <hip/hip_runtime.h>

// ---------------- problem constants ----------------
#define B_DIM   8
#define L_DIM   13
#define T_DIM   1500
#define D_DIM   768
#define K_MASK  750            // T * MASK_RATIO
#define TILES_T 94             // ceil(1500/16)
#define KSTEPS  24             // 768 / 32  (bf16 WMMA K=32)
#define NT      8              // N-tiles register-blocked per wave
#define NGROUPS 6              // 48 N-tiles / NT
#define WAVES_PER_WG 4
#define GEMM_BLOCK   (WAVES_PER_WG * 32)
#define LDS_UINTS_PER_WAVE (KSTEPS * 32 * 8)   // 24 KB per wave (A fragments)

typedef __attribute__((ext_vector_type(16))) __bf16        v16bf;
typedef __attribute__((ext_vector_type(8)))  float         v8f;
typedef __attribute__((ext_vector_type(4)))  unsigned int  u32x4;

struct ABits { u32x4 lo, hi; };   // 32 bytes == one v16bf fragment

// fp32 -> bf16 round-to-nearest-even
static __device__ __forceinline__ unsigned short f2bf(float f) {
    unsigned int u = __builtin_bit_cast(unsigned int, f);
    unsigned int r = u + 0x7FFFu + ((u >> 16) & 1u);
    return (unsigned short)(r >> 16);
}
static __device__ __forceinline__ unsigned int pack2(float lo, float hi) {
    return (unsigned int)f2bf(lo) | ((unsigned int)f2bf(hi) << 16);
}

// ---------------- kernel 0: convert W1 and Wf to bf16 ----------------
__global__ void cvt_weights(const float* __restrict__ W1, const float* __restrict__ Wf,
                            unsigned short* __restrict__ W1b, unsigned short* __restrict__ Wfb) {
    const size_t nW1 = (size_t)L_DIM * D_DIM * D_DIM;
    const size_t nWf = (size_t)D_DIM * D_DIM;
    size_t idx = (size_t)blockIdx.x * blockDim.x + threadIdx.x;
    if (idx < nW1)            W1b[idx] = f2bf(W1[idx]);
    else if (idx < nW1 + nWf) Wfb[idx - nW1] = f2bf(Wf[idx - nW1]);
}

// ---------------- unified attentive-score GEMM kernel ----------------
// scores[slab][t] = v[l] . tanh( X[slab][t][:] @ W[l]^T + bias[l] ) + c[l]
// X:[slabs,T,D] fp32, W:[Lmod,D,D] bf16 (row = out dim e, col = in dim d),
// slab -> l = slab % Lmod.  One wave per 16-frame tile, NT=8 N-tiles blocked.
__global__ void aspm_score_gemm(const float* __restrict__ X,
                                const unsigned short* __restrict__ Wb,
                                const float* __restrict__ bias,
                                const float* __restrict__ vvec,
                                const float* __restrict__ cvec,
                                float* __restrict__ scores,
                                int slabs, int Lmod) {
    extern __shared__ unsigned int ldsU[];
    const int wave = threadIdx.x >> 5;
    const int lane = threadIdx.x & 31;
    const int tile = blockIdx.x * WAVES_PER_WG + wave;
    const bool valid = tile < slabs * TILES_T;

    int slab = 0, t0 = 0, l = 0;
    if (valid) {
        slab = tile / TILES_T;
        t0   = (tile % TILES_T) * 16;
        l    = slab % Lmod;
    }
    unsigned int* myLds = ldsU + wave * LDS_UINTS_PER_WAVE;

    // ---- stage A tile (16 frames x 768 dims) into LDS in WMMA A-fragment order ----
    // A layout (16-bit, 16x32): lane holds row m = lane%16; half = lane/16.
    // frag elem e: d = kk*32 + 16*half + e         (e < 16)
    if (valid) {
        const int m = lane & 15, half = lane >> 4;
        int t = t0 + m; if (t > T_DIM - 1) t = T_DIM - 1;   // clamp padded rows
        const float* xr = X + (size_t)slab * T_DIM * D_DIM + (size_t)t * D_DIM;
        for (int kk = 0; kk < KSTEPS; ++kk) {
            const int d0 = kk * 32 + half * 16;
            float4 f0 = *(const float4*)(xr + d0);
            float4 f1 = *(const float4*)(xr + d0 + 4);
            float4 g0 = *(const float4*)(xr + d0 + 8);
            float4 g1 = *(const float4*)(xr + d0 + 12);
            uint4 lo = make_uint4(pack2(f0.x, f0.y), pack2(f0.z, f0.w),
                                  pack2(f1.x, f1.y), pack2(f1.z, f1.w));
            uint4 hi = make_uint4(pack2(g0.x, g0.y), pack2(g0.z, g0.w),
                                  pack2(g1.x, g1.y), pack2(g1.z, g1.w));
            uint4* p = (uint4*)(myLds + (kk * 32 + lane) * 8);
            p[0] = lo; p[1] = hi;
        }
    }
    __syncthreads();
    if (!valid) return;

    const int nl = lane & 15, half = lane >> 4;
    float acc[8];
#pragma unroll
    for (int r = 0; r < 8; ++r) acc[r] = 0.0f;

    const unsigned short* Wl = Wb + (size_t)l * D_DIM * D_DIM;
    // B layout (16-bit, 32x16): lane holds col e = tile_n*16 + nl,
    // K contiguous per fragment: k = kk*32 + 16*half + e
    const unsigned short* wbase = Wl + (size_t)nl * D_DIM + half * 16;
    // LDS byte address of this wave's fragment region (generic ptr low 32 bits
    // are the LDS byte offset per the ISA aperture rules)
    const unsigned ldsBase = (unsigned)(unsigned long long)(uintptr_t)myLds;

#pragma clang loop unroll(disable)
    for (int ng = 0; ng < NGROUPS; ++ng) {
        v8f C[NT];
#pragma unroll
        for (int j = 0; j < NT; ++j) C[j] = v8f{};
        const unsigned short* wg = wbase + (size_t)(ng * NT * 16) * D_DIM;

#pragma clang loop unroll(disable)
        for (int kk = 0; kk < KSTEPS; ++kk) {
            // A fragment from LDS; volatile asm so it cannot be hoisted/spilled
            const unsigned a32 = ldsBase + (unsigned)((kk * 32 + lane) * 32);
            u32x4 alo, ahi;
            asm volatile("ds_load_b128 %0, %2\n\t"
                         "ds_load_b128 %1, %2 offset:16\n\t"
                         "s_wait_dscnt 0"
                         : "=v"(alo), "=v"(ahi)
                         : "v"(a32));
            ABits abits{alo, ahi};
            v16bf a = __builtin_bit_cast(v16bf, abits);

            v16bf b[NT];
#pragma unroll
            for (int j = 0; j < NT; ++j)
                b[j] = *(const v16bf*)(wg + (size_t)(j * 16) * D_DIM + kk * 32);
#pragma unroll
            for (int j = 0; j < NT; ++j)
                C[j] = __builtin_amdgcn_wmma_f32_16x16x32_bf16(
                           false, a, false, b[j], (short)0, C[j], false, false);
        }

        // epilogue: bias + tanh + dot with v, folded into per-lane accumulators
        // C layout: lane holds col e; VGPR r holds row m = r + 8*half
#pragma unroll
        for (int j = 0; j < NT; ++j) {
            const int e = (ng * NT + j) * 16 + nl;
            const float bi = bias[(size_t)l * D_DIM + e];
            const float vv = vvec[(size_t)l * D_DIM + e];
#pragma unroll
            for (int r = 0; r < 8; ++r) acc[r] += vv * tanhf(C[j][r] + bi);
        }
    }

    // reduce over e across the 16 lanes of each half-wave
#pragma unroll
    for (int off = 1; off < 16; off <<= 1) {
#pragma unroll
        for (int r = 0; r < 8; ++r) acc[r] += __shfl_xor(acc[r], off, 32);
    }

    if (nl == 0) {  // lane 0 -> rows 0..7, lane 16 -> rows 8..15
        const float c = cvec[l];
        float* so = scores + (size_t)slab * T_DIM + t0 + half * 8;
#pragma unroll
        for (int r = 0; r < 8; ++r) {
            int m = half * 8 + r;
            if (t0 + m < T_DIM) so[r] = acc[r] + c;
        }
    }
}

// ---------------- kernel 2: stable rank mask + softmax over T ----------------
__global__ void rank_mask_softmax(const float* __restrict__ scores,
                                  float* __restrict__ attn) {
    __shared__ float s[T_DIM];
    __shared__ float ms[T_DIM];
    __shared__ float red[256];
    const int slab = blockIdx.x;
    const float* sp = scores + (size_t)slab * T_DIM;
    for (int i = threadIdx.x; i < T_DIM; i += blockDim.x) s[i] = sp[i];
    __syncthreads();
    // stable double-argsort rank: #(s_j < s_i) + #(s_j == s_i && j < i)
    for (int i = threadIdx.x; i < T_DIM; i += blockDim.x) {
        float si = s[i];
        int rank = 0;
        for (int j = 0; j < T_DIM; ++j) {
            float sj = s[j];
            rank += (sj < si) || (sj == si && j < i);
        }
        ms[i] = (rank >= K_MASK) ? si : -__builtin_inff();
    }
    __syncthreads();
    float m = -__builtin_inff();
    for (int i = threadIdx.x; i < T_DIM; i += blockDim.x) m = fmaxf(m, ms[i]);
    red[threadIdx.x] = m; __syncthreads();
    for (int o = 128; o > 0; o >>= 1) {
        if (threadIdx.x < o) red[threadIdx.x] = fmaxf(red[threadIdx.x], red[threadIdx.x + o]);
        __syncthreads();
    }
    m = red[0]; __syncthreads();
    float sum = 0.0f;
    for (int i = threadIdx.x; i < T_DIM; i += blockDim.x) {
        float e = expf(ms[i] - m);
        ms[i] = e; sum += e;
    }
    red[threadIdx.x] = sum; __syncthreads();
    for (int o = 128; o > 0; o >>= 1) {
        if (threadIdx.x < o) red[threadIdx.x] += red[threadIdx.x + o];
        __syncthreads();
    }
    const float inv = 1.0f / red[0];
    float* ap = attn + (size_t)slab * T_DIM;
    for (int i = threadIdx.x; i < T_DIM; i += blockDim.x) ap[i] = ms[i] * inv;
}

// ---------------- kernel 3: layer-weighted combine -> weighted[B,T,D] ----------------
__global__ void layer_combine(const float* __restrict__ reps,
                              const float* __restrict__ attn,
                              const float* __restrict__ lw,
                              float* __restrict__ weighted) {
    const int DV = D_DIM / 4;
    size_t idx = (size_t)blockIdx.x * blockDim.x + threadIdx.x;
    size_t total = (size_t)B_DIM * T_DIM * DV;
    if (idx >= total) return;
    int dv = (int)(idx % DV);
    size_t bt = idx / DV;
    int t = (int)(bt % T_DIM);
    int b = (int)(bt / T_DIM);
    float ax = 0.f, ay = 0.f, az = 0.f, aw_ = 0.f;
#pragma unroll
    for (int l = 0; l < L_DIM; ++l) {
        const size_t sl = (size_t)b * L_DIM + l;
        const float w = attn[sl * T_DIM + t] * lw[l];
        const float4 x = *(const float4*)(reps + (sl * T_DIM + t) * D_DIM + dv * 4);
        ax += x.x * w; ay += x.y * w; az += x.z * w; aw_ += x.w * w;
    }
    *(float4*)(weighted + ((size_t)b * T_DIM + t) * D_DIM + dv * 4) =
        make_float4(ax, ay, az, aw_);
}

// ---------------- kernel 5: row softmax (per batch over T) ----------------
__global__ void softmax_rows(const float* __restrict__ s, float* __restrict__ aw) {
    __shared__ float ms[T_DIM];
    __shared__ float red[256];
    const int b = blockIdx.x;
    const float* sp = s + (size_t)b * T_DIM;
    float m = -__builtin_inff();
    for (int i = threadIdx.x; i < T_DIM; i += blockDim.x) {
        float v = sp[i]; ms[i] = v; m = fmaxf(m, v);
    }
    red[threadIdx.x] = m; __syncthreads();
    for (int o = 128; o > 0; o >>= 1) {
        if (threadIdx.x < o) red[threadIdx.x] = fmaxf(red[threadIdx.x], red[threadIdx.x + o]);
        __syncthreads();
    }
    m = red[0]; __syncthreads();
    float sum = 0.0f;
    for (int i = threadIdx.x; i < T_DIM; i += blockDim.x) {
        float e = expf(ms[i] - m);
        ms[i] = e; sum += e;
    }
    red[threadIdx.x] = sum; __syncthreads();
    for (int o = 128; o > 0; o >>= 1) {
        if (threadIdx.x < o) red[threadIdx.x] += red[threadIdx.x + o];
        __syncthreads();
    }
    const float inv = 1.0f / red[0];
    float* ap = aw + (size_t)b * T_DIM;
    for (int i = threadIdx.x; i < T_DIM; i += blockDim.x) ap[i] = ms[i] * inv;
}

// ---------------- kernel 6: final pooling out[b,d] = sum_t aw[b,t]*weighted[b,t,d] ----
__global__ void pool_out(const float* __restrict__ weighted,
                         const float* __restrict__ aw,
                         float* __restrict__ out) {
    __shared__ float a[T_DIM];
    const int b = blockIdx.y;
    const int d = blockIdx.x * blockDim.x + threadIdx.x;
    for (int i = threadIdx.x; i < T_DIM; i += blockDim.x) a[i] = aw[(size_t)b * T_DIM + i];
    __syncthreads();
    float acc = 0.0f;
    const float* wp = weighted + (size_t)b * T_DIM * D_DIM + d;
    for (int t = 0; t < T_DIM; ++t) acc += a[t] * wp[(size_t)t * D_DIM];
    out[(size_t)b * D_DIM + d] = acc;
}

// ---------------- host launcher ----------------
static inline size_t align256(size_t x) { return (x + 255) & ~(size_t)255; }

extern "C" void kernel_launch(void* const* d_in, const int* in_sizes, int n_in,
                              void* d_out, int out_size, void* d_ws, size_t ws_size,
                              hipStream_t stream) {
    const float* reps = (const float*)d_in[0];   // [B,L,T,D]
    const float* W1   = (const float*)d_in[1];   // [L,D,D]
    const float* b1   = (const float*)d_in[2];   // [L,D]
    const float* v1   = (const float*)d_in[3];   // [L,D]
    const float* c1   = (const float*)d_in[4];   // [L]
    const float* lw   = (const float*)d_in[5];   // [L]
    const float* Wf   = (const float*)d_in[6];   // [D,D]
    const float* bf   = (const float*)d_in[7];   // [D]
    const float* vf   = (const float*)d_in[8];   // [D]
    const float* cf   = (const float*)d_in[9];   // scalar
    float* out = (float*)d_out;                  // [B,D]

    char* ws = (char*)d_ws;
    size_t off = 0;
    unsigned short* W1b = (unsigned short*)(ws + off); off += align256((size_t)L_DIM * D_DIM * D_DIM * 2);
    unsigned short* Wfb = (unsigned short*)(ws + off); off += align256((size_t)D_DIM * D_DIM * 2);
    float* scores   = (float*)(ws + off); off += align256((size_t)B_DIM * L_DIM * T_DIM * 4);
    float* attn     = (float*)(ws + off); off += align256((size_t)B_DIM * L_DIM * T_DIM * 4);
    float* weighted = (float*)(ws + off); off += align256((size_t)B_DIM * T_DIM * D_DIM * 4);
    float* sfinal   = (float*)(ws + off); off += align256((size_t)B_DIM * T_DIM * 4);
    float* aw       = (float*)(ws + off); off += align256((size_t)B_DIM * T_DIM * 4);
    (void)ws_size; (void)in_sizes; (void)n_in; (void)out_size;

    // 0) weights -> bf16
    {
        size_t n = (size_t)L_DIM * D_DIM * D_DIM + (size_t)D_DIM * D_DIM;
        int g = (int)((n + 255) / 256);
        cvt_weights<<<g, 256, 0, stream>>>(W1, Wf, W1b, Wfb);
    }
    // 1) per-layer ASPM scores (bf16 WMMA)
    {
        int slabs = B_DIM * L_DIM;
        int tiles = slabs * TILES_T;
        int g = (tiles + WAVES_PER_WG - 1) / WAVES_PER_WG;
        size_t sh = (size_t)WAVES_PER_WG * LDS_UINTS_PER_WAVE * 4;
        aspm_score_gemm<<<g, GEMM_BLOCK, sh, stream>>>(
            reps, W1b, b1, v1, c1, scores, slabs, L_DIM);
    }
    // 2) rank mask + softmax per (b,l)
    rank_mask_softmax<<<B_DIM * L_DIM, 256, 0, stream>>>(scores, attn);
    // 3) layer combine -> weighted
    {
        size_t n = (size_t)B_DIM * T_DIM * (D_DIM / 4);
        int g = (int)((n + 255) / 256);
        layer_combine<<<g, 256, 0, stream>>>(reps, attn, lw, weighted);
    }
    // 4) final attention scores (same WMMA kernel, Lmod=1)
    {
        int slabs = B_DIM;
        int tiles = slabs * TILES_T;
        int g = (tiles + WAVES_PER_WG - 1) / WAVES_PER_WG;
        size_t sh = (size_t)WAVES_PER_WG * LDS_UINTS_PER_WAVE * 4;
        aspm_score_gemm<<<g, GEMM_BLOCK, sh, stream>>>(
            weighted, Wfb, bf, vf, cf, sfinal, slabs, 1);
    }
    // 5) softmax over frames per batch
    softmax_rows<<<B_DIM, 256, 0, stream>>>(sfinal, aw);
    // 6) attentive pooling -> out[B,D]
    {
        dim3 g(D_DIM / 256, B_DIM);
        pool_out<<<g, 256, 0, stream>>>(weighted, aw, out);
    }
}
